// GlmAsrAttention_26036091748759
// MI455X (gfx1250) — compile-verified, matrix operand order checked by reading, not run on hardware
//
#include <hip/hip_runtime.h>
#include <stdint.h>

// ---------------- problem constants ----------------
#define Bb 4
#define Ss 2048
#define Hh 2048
#define NH 16
#define NKV 4
#define HD 128
#define QKVC 3072           // (NH + 2*NKV)*HD
#define MROWS (Bb*Ss)       // 8192

typedef __bf16 bf16;
typedef __bf16 v16bf __attribute__((ext_vector_type(16)));
typedef float  v8f   __attribute__((ext_vector_type(8)));
typedef int    b128v __attribute__((vector_size(16)));   // matches builtin param type

// CDNA5 async global->LDS path (ASYNCcnt), guarded so either toolchain compiles.
#if defined(__gfx1250__) && __has_builtin(__builtin_amdgcn_global_load_async_to_lds_b128)
#define HAVE_ASYNC_LDS 1
#else
#define HAVE_ASYNC_LDS 0
#endif

// Generic -> explicit address-space casts with the builtin's exact pointee type.
// Global aperture: generic value == global address; LDS aperture: low 32 bits of
// the generic address are the LDS offset.
#define AS1(p) ((__attribute__((address_space(1))) b128v*)(uintptr_t)(p))
#define AS3(p) ((__attribute__((address_space(3))) b128v*)(uint32_t)(uintptr_t)(p))

// ---------------- helpers ----------------
__device__ __forceinline__ bf16 f2bf(float f) {
  union { float f; uint32_t u; } x; x.f = f;
  uint32_t r = (x.u + 0x7FFFu + ((x.u >> 16) & 1u)) >> 16;
  union { uint16_t s; bf16 b; } y; y.s = (uint16_t)r;
  return y.b;
}

__device__ __forceinline__ v8f wmma_bf16(v16bf a, v16bf b, v8f c) {
  // emits v_wmma_f32_16x16x32_bf16
  return __builtin_amdgcn_wmma_f32_16x16x32_bf16(false, a, false, b, (short)0, c, false, false);
}

// Load a 16x32 A/B fragment (bf16) from a row-major buffer.
// Per CDNA5 ISA 7.12.2 (16-bit A-matrix 16x32): lane m = lane&15,
// lanes 0-15 hold K = {0..7,16..23}, lanes 16-31 hold K = {8..15,24..31};
// vector elem e<8 -> K=kb+e, e>=8 -> K=kb+16+(e-8).
__device__ __forceinline__ v16bf frag16(const bf16* base, int row0, int ld, int lane) {
  int m  = lane & 15;
  int kb = (lane >> 4) << 3;
  const bf16* p = base + (size_t)(row0 + m) * ld + kb;
  v16bf f;
#pragma unroll
  for (int e = 0; e < 8; ++e) { f[e] = p[e]; f[e + 8] = p[e + 16]; }
  return f;
}

// ---------------- cast kernels ----------------
__global__ void cast_bf16_kernel(const float* __restrict__ src, bf16* __restrict__ dst, int n) {
  int i = blockIdx.x * 256 + threadIdx.x;
  if (i < n) dst[i] = f2bf(src[i]);
}

// dst[c][r] = bf16(src[r][c]) ; src is R x C row-major, dst is C x R row-major
__global__ void transpose_cast_kernel(const float* __restrict__ src, bf16* __restrict__ dst,
                                      int R, int C) {
  __shared__ float tile[32][33];
  int c0 = blockIdx.x * 32, r0 = blockIdx.y * 32;
  int tx = threadIdx.x, ty = threadIdx.y;   // 32 x 8
#pragma unroll
  for (int i = 0; i < 32; i += 8)
    tile[ty + i][tx] = src[(size_t)(r0 + ty + i) * C + c0 + tx];
  __syncthreads();
#pragma unroll
  for (int i = 0; i < 32; i += 8)
    dst[(size_t)(c0 + ty + i) * R + r0 + tx] = f2bf(tile[tx][ty + i]);
}

// ---------------- WMMA GEMM:  C[M,N] = A[M,K](bf16) * Bt[N,K](bf16)^T + bias ----------------
// 128 threads = 4 waves, 64x64 block tile, 32x32 per wave (2x2 WMMA tiles),
// double-buffered LDS with async global->LDS staging of the k+1 tile.
__global__ __launch_bounds__(128)
void gemm_bf16_bias_kernel(const bf16* __restrict__ A, const bf16* __restrict__ Bt,
                           const float* __restrict__ bias, float* __restrict__ C,
                           int M, int N, int K) {
  __shared__ alignas(16) bf16 As[2][64 * 32];
  __shared__ alignas(16) bf16 Bs[2][64 * 32];
  const int n0 = blockIdx.x * 64;
  const int m0 = blockIdx.y * 64;
  const int t = threadIdx.x;
  const int wave = t >> 5, lane = t & 31;
  const int wm = (wave >> 1) * 32;
  const int wn = (wave & 1) * 32;

  auto stage = [&](int buf, int k0) {
#if HAVE_ASYNC_LDS
    // 64x32 bf16 = 4 KiB per tile = 256 b128 chunks; 2 per thread per matrix.
#pragma unroll
    for (int ii = 0; ii < 2; ++ii) {
      int i = t + ii * 128;
      int r = i >> 2, c = (i & 3) * 8;
      __builtin_amdgcn_global_load_async_to_lds_b128(
          AS1(A + (size_t)(m0 + r) * K + k0 + c), AS3(&As[buf][r * 32 + c]), 0, 0);
      __builtin_amdgcn_global_load_async_to_lds_b128(
          AS1(Bt + (size_t)(n0 + r) * K + k0 + c), AS3(&Bs[buf][r * 32 + c]), 0, 0);
    }
#else
#pragma unroll
    for (int ii = 0; ii < 8; ++ii) {
      int i = t + ii * 128;
      int r = i >> 4, cw = i & 15;
      ((uint32_t*)As[buf])[i] = *(const uint32_t*)(A  + (size_t)(m0 + r) * K + k0 + cw * 2);
      ((uint32_t*)Bs[buf])[i] = *(const uint32_t*)(Bt + (size_t)(n0 + r) * K + k0 + cw * 2);
    }
#endif
  };

  v8f acc[2][2];
#pragma unroll
  for (int a = 0; a < 2; ++a)
#pragma unroll
    for (int b = 0; b < 2; ++b)
#pragma unroll
      for (int e = 0; e < 8; ++e) acc[a][b][e] = 0.0f;

  stage(0, 0);
#if HAVE_ASYNC_LDS
  asm volatile("s_wait_asynccnt 0x0" ::: "memory");
#endif
  __syncthreads();

  int cur = 0;
  for (int k0 = 0; k0 < K; k0 += 32) {
    if (k0 + 32 < K) stage(cur ^ 1, k0 + 32);   // prefetch next tile (async)

    v16bf af0 = frag16(&As[cur][0], wm,      32, lane);
    v16bf af1 = frag16(&As[cur][0], wm + 16, 32, lane);
    v16bf bf0 = frag16(&Bs[cur][0], wn,      32, lane);
    v16bf bf1 = frag16(&Bs[cur][0], wn + 16, 32, lane);
    acc[0][0] = wmma_bf16(af0, bf0, acc[0][0]);
    acc[0][1] = wmma_bf16(af0, bf1, acc[0][1]);
    acc[1][0] = wmma_bf16(af1, bf0, acc[1][0]);
    acc[1][1] = wmma_bf16(af1, bf1, acc[1][1]);

#if HAVE_ASYNC_LDS
    asm volatile("s_wait_asynccnt 0x0" ::: "memory");
#endif
    __syncthreads();
    cur ^= 1;
  }

  // epilogue: C/D layout — row = i + 8*(lane>>4), col = lane&15
  const int colb = lane & 15;
  const int rhi  = (lane >> 4) * 8;
#pragma unroll
  for (int mi = 0; mi < 2; ++mi)
#pragma unroll
    for (int ni = 0; ni < 2; ++ni)
#pragma unroll
      for (int i = 0; i < 8; ++i) {
        int r = m0 + wm + mi * 16 + rhi + i;
        int c = n0 + wn + ni * 16 + colb;
        C[(size_t)r * N + c] = acc[mi][ni][i] + bias[c];
      }
}

// ---------------- RoPE + bf16 cast of QKV ----------------
// qkv rows = b*S+s, cols: [0,2048) q (rope), [2048,2560) k (rope), [2560,3072) v (copy)
__global__ void rope_cast_kernel(const float* __restrict__ qkv, const float* __restrict__ cosb,
                                 const float* __restrict__ sinb, bf16* __restrict__ out) {
  size_t idx = (size_t)blockIdx.x * 256 + threadIdx.x;   // MROWS*QKVC total
  int col = (int)(idx % QKVC);
  size_t row = idx / QKVC;
  float x = qkv[idx];
  float y;
  if (col < 2560) {                       // q or k head -> rope (ROT == HD == 128)
    int d = col & 127;
    float c = cosb[row * 128 + d];
    float s = sinb[row * 128 + d];
    float xr = qkv[(d < 64) ? (idx + 64) : (idx - 64)];
    y = x * c + ((d < 64) ? -xr : xr) * s;
  } else {
    y = x;
  }
  out[idx] = f2bf(y);
}

// ---------------- flash attention (bf16 WMMA, fp32 online softmax) ----------------
// grid = B*NH*(S/128); 256 threads = 8 waves, each wave owns 16 query rows.
// K tile async-staged row-major; V tile staged transposed; double-buffered.
__global__ __launch_bounds__(256)
void attn_kernel(const bf16* __restrict__ qkv, bf16* __restrict__ attn) {
  __shared__ alignas(16) bf16 Kt[2][32 * 128];   // kv-rows x d (row-major)
  __shared__ alignas(16) bf16 Vt[2][128 * 32];   // d x kv-rows (transposed for B-fragments)
  __shared__ alignas(16) bf16 Pst[8][16 * 32];   // per-wave P turn-around (C->A layout)

  const int bid = blockIdx.x;
  const int qt  = bid & 15;          // S/128 = 16 query tiles
  const int h   = (bid >> 4) & 15;   // NH
  const int b   = bid >> 8;          // B
  const int kvh = h >> 2;            // GQA group
  const int t = threadIdx.x;
  const int wave = t >> 5, lane = t & 31;

  const size_t rowQ0  = (size_t)b * Ss + qt * 128 + wave * 16;
  const size_t kvbase = (size_t)b * Ss;

  auto stage = [&](int buf, int j0) {
#if HAVE_ASYNC_LDS
    // K tile: 32x128 bf16 = 8 KiB = 512 b128 chunks; 2 per thread, direct to LDS.
#pragma unroll
    for (int ii = 0; ii < 2; ++ii) {
      int i = t + ii * 256;
      int kk = i >> 4, c = (i & 15) * 8;
      __builtin_amdgcn_global_load_async_to_lds_b128(
          AS1(qkv + (kvbase + j0 + kk) * QKVC + 2048 + kvh * HD + c),
          AS3(&Kt[buf][kk * 128 + c]), 0, 0);
    }
    // V tile transposed: element scatter (sync path)
#pragma unroll
    for (int ii = 0; ii < 16; ++ii) {
      int i = t + ii * 256;
      int kk = i >> 7, d = i & 127;
      Vt[buf][d * 32 + kk] = qkv[(kvbase + j0 + kk) * QKVC + 2560 + kvh * HD + d];
    }
#else
#pragma unroll
    for (int ii = 0; ii < 16; ++ii) {
      int i = t + ii * 256;
      int kk = i >> 7, d = i & 127;
      size_t r = (kvbase + j0 + kk) * QKVC;
      Kt[buf][kk * 128 + d] = qkv[r + 2048 + kvh * HD + d];
      Vt[buf][d * 32 + kk]  = qkv[r + 2560 + kvh * HD + d];
    }
#endif
  };

  // Q fragments for this wave's 16 rows (HD=128 -> 4 K-chunks of 32)
  v16bf qf[4];
  {
    int m = lane & 15, kb = (lane >> 4) << 3;
    const bf16* qp = qkv + (rowQ0 + m) * QKVC + h * HD;
#pragma unroll
    for (int kc = 0; kc < 4; ++kc) {
      const bf16* p = qp + kc * 32 + kb;
#pragma unroll
      for (int e = 0; e < 8; ++e) { qf[kc][e] = p[e]; qf[kc][e + 8] = p[e + 16]; }
    }
  }

  float mstat[8], lstat[8];
  v8f o[8];
#pragma unroll
  for (int i = 0; i < 8; ++i) {
    mstat[i] = -3.0e38f; lstat[i] = 0.0f;
#pragma unroll
    for (int e = 0; e < 8; ++e) o[i][e] = 0.0f;
  }

  const float scale = 0.08838834764831845f;   // 1/sqrt(128)

  stage(0, 0);
#if HAVE_ASYNC_LDS
  asm volatile("s_wait_asynccnt 0x0" ::: "memory");
#endif
  __syncthreads();

  int cur = 0;
  for (int j0 = 0; j0 < Ss; j0 += 32) {
    if (j0 + 32 < Ss) stage(cur ^ 1, j0 + 32);   // prefetch next KV tile

    // scores: S = Q(16x128) * K_tile^T(128x32) -> two 16x16 f32 tiles
    v8f sc[2];
#pragma unroll
    for (int n = 0; n < 2; ++n)
#pragma unroll
      for (int e = 0; e < 8; ++e) sc[n][e] = 0.0f;
#pragma unroll
    for (int nch = 0; nch < 2; ++nch)
#pragma unroll
      for (int kc = 0; kc < 4; ++kc) {
        v16bf kf = frag16(&Kt[cur][0] + kc * 32, nch * 16, 128, lane);
        sc[nch] = wmma_bf16(qf[kc], kf, sc[nch]);
      }

    // online softmax update (row r = i + 8*(lane>>4); each row spans 16 lanes)
    float alpha[8];
#pragma unroll
    for (int i = 0; i < 8; ++i) {
      float s0 = sc[0][i] * scale;
      float s1 = sc[1][i] * scale;
      float mx = fmaxf(s0, s1);
      mx = fmaxf(mx, __shfl_xor(mx, 1, 32));
      mx = fmaxf(mx, __shfl_xor(mx, 2, 32));
      mx = fmaxf(mx, __shfl_xor(mx, 4, 32));
      mx = fmaxf(mx, __shfl_xor(mx, 8, 32));
      float nm = fmaxf(mstat[i], mx);
      float p0 = __expf(s0 - nm);
      float p1 = __expf(s1 - nm);
      float rs = p0 + p1;
      rs += __shfl_xor(rs, 1, 32);
      rs += __shfl_xor(rs, 2, 32);
      rs += __shfl_xor(rs, 4, 32);
      rs += __shfl_xor(rs, 8, 32);
      float a = __expf(mstat[i] - nm);
      mstat[i] = nm;
      lstat[i] = lstat[i] * a + rs;
      alpha[i] = a;
      sc[0][i] = p0;
      sc[1][i] = p1;
    }
#pragma unroll
    for (int dch = 0; dch < 8; ++dch)
#pragma unroll
      for (int i = 0; i < 8; ++i) o[dch][i] *= alpha[i];

    // P (C-layout) -> LDS -> A-fragment layout, wave-private region
    {
      bf16* P = &Pst[wave][0];
      int colb = lane & 15, rhi = (lane >> 4) * 8;
#pragma unroll
      for (int i = 0; i < 8; ++i) {
        P[(rhi + i) * 32 + colb]      = f2bf(sc[0][i]);
        P[(rhi + i) * 32 + 16 + colb] = f2bf(sc[1][i]);
      }
    }
    asm volatile("s_wait_dscnt 0x0" ::: "memory");   // CDNA5 split DS counter

    // O += P(16x32) * V_tile(32x128)
    {
      v16bf pf = frag16(&Pst[wave][0], 0, 32, lane);
#pragma unroll
      for (int dch = 0; dch < 8; ++dch) {
        v16bf vf = frag16(&Vt[cur][0], dch * 16, 32, lane);
        o[dch] = wmma_bf16(pf, vf, o[dch]);
      }
    }

#if HAVE_ASYNC_LDS
    asm volatile("s_wait_asynccnt 0x0" ::: "memory");
#endif
    __syncthreads();   // buffer swap point
    cur ^= 1;
  }

  // finalize: O/l -> bf16 into (B*S, NH*HD)
  {
    int colb = lane & 15, rhi = (lane >> 4) * 8;
#pragma unroll
    for (int i = 0; i < 8; ++i) {
      float inv = 1.0f / lstat[i];
      size_t row = rowQ0 + rhi + i;
#pragma unroll
      for (int dch = 0; dch < 8; ++dch)
        attn[row * (NH * HD) + h * HD + dch * 16 + colb] = f2bf(o[dch][i] * inv);
    }
  }
}

// ---------------- launch ----------------
extern "C" void kernel_launch(void* const* d_in, const int* in_sizes, int n_in,
                              void* d_out, int out_size, void* d_ws, size_t ws_size,
                              hipStream_t stream) {
  const float* hidden = (const float*)d_in[0];
  const float* cosb   = (const float*)d_in[1];
  const float* sinb   = (const float*)d_in[2];
  const float* w_qkv  = (const float*)d_in[3];
  const float* b_qkv  = (const float*)d_in[4];
  const float* w_o    = (const float*)d_in[5];
  const float* b_o    = (const float*)d_in[6];
  float* out = (float*)d_out;

  char* ws = (char*)d_ws;
  bf16*  hidden_bf = (bf16*)(ws);                                   // 32 MiB
  bf16*  wqkvT     = (bf16*)(ws + (size_t)33554432);                // 12 MiB
  bf16*  woT       = (bf16*)(ws + (size_t)46137344);                // 8 MiB
  float* qkv_f     = (float*)(ws + (size_t)54525952);               // 96 MiB
  bf16*  qkv_bf    = (bf16*)(ws + (size_t)155189248);               // 48 MiB
  bf16*  attn_bf   = (bf16*)(ws + (size_t)205520896);               // 32 MiB

  // 1) casts / weight transposes
  cast_bf16_kernel<<<(MROWS * Hh) / 256, 256, 0, stream>>>(hidden, hidden_bf, MROWS * Hh);
  dim3 tb(32, 8);
  transpose_cast_kernel<<<dim3(QKVC / 32, Hh / 32), tb, 0, stream>>>(w_qkv, wqkvT, Hh, QKVC);
  transpose_cast_kernel<<<dim3(Hh / 32, Hh / 32),   tb, 0, stream>>>(w_o,   woT,   Hh, Hh);

  // 2) QKV projection (WMMA)
  gemm_bf16_bias_kernel<<<dim3(QKVC / 64, MROWS / 64), 128, 0, stream>>>(
      hidden_bf, wqkvT, b_qkv, qkv_f, MROWS, QKVC, Hh);

  // 3) RoPE + bf16 cast
  rope_cast_kernel<<<(MROWS * QKVC) / 256, 256, 0, stream>>>(qkv_f, cosb, sinb, qkv_bf);

  // 4) flash attention (WMMA)
  attn_kernel<<<Bb * NH * (Ss / 128), 256, 0, stream>>>(qkv_bf, attn_bf);

  // 5) output projection (WMMA) -> fp32 out
  gemm_bf16_bias_kernel<<<dim3(Hh / 64, MROWS / 64), 128, 0, stream>>>(
      attn_bf, woT, b_o, out, MROWS, Hh, Hh);
}
